// WordNLM_59889023975695
// MI455X (gfx1250) — compile-verified, matrix operand order checked by reading, not run on hardware
//
#include <hip/hip_runtime.h>

typedef unsigned short u16;
typedef unsigned int   u32;
typedef __attribute__((ext_vector_type(16))) __bf16 v16bf;
typedef __attribute__((ext_vector_type(8)))  float  v8f;

#define SEQ    128
#define BATCH  16
#define EMBED  1024
#define HIDDEN 1024
#define GATES  4096          // 4*HIDDEN
#define VOCAB  50257
#define MROWS  2048          // SEQ*BATCH
#define LPAD   40            // LDS row stride (elems): 80B, bank-friendly

// ---------- helpers ----------

__device__ __forceinline__ u16 f2bf(float f) {
    u32 u = __float_as_uint(f);
    u32 r = (u + 0x7FFFu + ((u >> 16) & 1u)) >> 16;   // RNE
    return (u16)r;
}

__device__ __forceinline__ float sigmoidf(float x) {
    return 1.0f / (1.0f + __expf(-x));
}

// Global-memory fragment load, CDNA5 16-bit A/B layout: lanes 0-15 hold
// K = [0..7],[16..23]; lanes 16-31 hold K = [8..15],[24..31] of row (lane&15).
__device__ __forceinline__ v16bf load_frag(const u16* __restrict__ p, int lane) {
    const u16* q = p + (((lane >> 4) & 1) * 8);
    union { v16bf v; uint4 u[2]; } f;
    f.u[0] = *(const uint4*)(q);
    f.u[1] = *(const uint4*)(q + 16);
    return f.v;
}

__device__ __forceinline__ v8f wmma_bf16(v16bf a, v16bf b, v8f c) {
    // D = A(16x32) * B(32x16) + C ; emits v_wmma_f32_16x16x32_bf16
    return __builtin_amdgcn_wmma_f32_16x16x32_bf16(
        false, a, false, b, (short)0, c, false, false);
}

// ---------- small utility kernels ----------

__global__ __launch_bounds__(256) void k_f32_to_bf16(const float* __restrict__ src,
                                                     u16* __restrict__ dst, int n) {
    for (int i = blockIdx.x * 256 + threadIdx.x; i < n; i += gridDim.x * 256)
        dst[i] = f2bf(src[i]);
}

__global__ __launch_bounds__(256) void k_embed_gather(const int* __restrict__ tokens,
                                                      const float* __restrict__ emb,
                                                      u16* __restrict__ x, int n) {
    for (int i = blockIdx.x * 256 + threadIdx.x; i < n; i += gridDim.x * 256) {
        int row = i >> 10;           // (t*BATCH + b)
        int e   = i & 1023;
        int tok = tokens[row];
        x[i] = f2bf(emb[(size_t)tok * EMBED + e]);
    }
}

__global__ __launch_bounds__(256) void k_bias_combine(const float* __restrict__ bi0,
                                                      const float* __restrict__ bh0,
                                                      const float* __restrict__ bi1,
                                                      const float* __restrict__ bh1,
                                                      float* __restrict__ o0,
                                                      float* __restrict__ o1) {
    int i = blockIdx.x * 256 + threadIdx.x;
    if (i < GATES) {
        o0[i] = bi0[i] + bh0[i];
        o1[i] = bi1[i] + bh1[i];
    }
}

__global__ __launch_bounds__(256) void k_zero_state(float* __restrict__ c,
                                                    u16* __restrict__ hzero) {
    int i = blockIdx.x * 256 + threadIdx.x;
    if (i < BATCH * HIDDEN) {
        c[i]     = 0.0f;
        hzero[i] = 0;
    }
}

// ---------- LDS-staged WMMA GEMM:  C[M,N] = A[M,K] * B[N,K]^T + bias[N] ----
// 256 threads = 8 waves arranged 4(M) x 2(N); workgroup tile 128x128.
// K is consumed in 32-wide slabs staged through double-buffered LDS tiles
// (128x32 for A and for B, rows padded to 80B). Per slab: issue next slab's
// global_load_b128s first, run 8 WMMAs from ds_load_b128 fragments, then
// ds_store the staged regs and one barrier. M % 128 == 0 assumed.
__global__ __launch_bounds__(256) void k_gemm_bf16(const u16* __restrict__ A,
                                                   const u16* __restrict__ B,
                                                   const float* __restrict__ bias,
                                                   float* __restrict__ C,
                                                   int N, int K) {
    __shared__ __align__(128) u16 sA[2][128][LPAD];
    __shared__ __align__(128) u16 sB[2][128][LPAD];

    const int tid  = threadIdx.x;
    const int lane = tid & 31;
    const int wave = tid >> 5;
    const int wm = wave & 3;              // 0..3 : which 32-row strip
    const int wn = wave >> 2;             // 0..1 : which 64-col strip
    const int m_blk = blockIdx.y * 128;
    const int n_blk = blockIdx.x * 128;

    // cooperative stage: each thread owns two 16B quads of each 128x32 tile
    const int q0 = tid * 2, q1 = q0 + 1;
    const int r0 = q0 >> 2, c0 = (q0 & 3) * 8;
    const int r1 = q1 >> 2, c1 = (q1 & 3) * 8;
    const u16* Ag0 = A + (size_t)(m_blk + r0) * K + c0;
    const u16* Ag1 = A + (size_t)(m_blk + r1) * K + c1;
    int bg0 = n_blk + r0; bg0 = (bg0 < N) ? bg0 : N - 1;   // ragged-N clamp
    int bg1 = n_blk + r1; bg1 = (bg1 < N) ? bg1 : N - 1;
    const u16* Bg0 = B + (size_t)bg0 * K + c0;
    const u16* Bg1 = B + (size_t)bg1 * K + c1;

    // prologue: stage slab 0
    {
        uint4 a0 = *(const uint4*)Ag0;
        uint4 a1 = *(const uint4*)Ag1;
        uint4 b0 = *(const uint4*)Bg0;
        uint4 b1 = *(const uint4*)Bg1;
        *(uint4*)&sA[0][r0][c0] = a0;
        *(uint4*)&sA[0][r1][c1] = a1;
        *(uint4*)&sB[0][r0][c0] = b0;
        *(uint4*)&sB[0][r1][c1] = b1;
    }
    __syncthreads();

    v8f acc[2][4] = {};
    const int fr   = lane & 15;           // fragment row within 16-row tile
    const int half = ((lane >> 4) & 1) * 8;

    int buf = 0;
    for (int k0 = 0; k0 < K; k0 += 32, buf ^= 1) {
        const bool has_next = (k0 + 32) < K;
        uint4 na0, na1, nb0, nb1;
        if (has_next) {                   // uniform branch: EXEC stays full
            na0 = *(const uint4*)(Ag0 + k0 + 32);
            na1 = *(const uint4*)(Ag1 + k0 + 32);
            nb0 = *(const uint4*)(Bg0 + k0 + 32);
            nb1 = *(const uint4*)(Bg1 + k0 + 32);
        }

        // fragments from LDS
        v16bf af[2], bf[4];
#pragma unroll
        for (int i = 0; i < 2; ++i) {
            union { v16bf v; uint4 u[2]; } f;
            const u16* p = &sA[buf][wm * 32 + i * 16 + fr][half];
            f.u[0] = *(const uint4*)(p);
            f.u[1] = *(const uint4*)(p + 16);
            af[i] = f.v;
        }
#pragma unroll
        for (int j = 0; j < 4; ++j) {
            union { v16bf v; uint4 u[2]; } f;
            const u16* p = &sB[buf][wn * 64 + j * 16 + fr][half];
            f.u[0] = *(const uint4*)(p);
            f.u[1] = *(const uint4*)(p + 16);
            bf[j] = f.v;
        }

#pragma unroll
        for (int i = 0; i < 2; ++i)
#pragma unroll
            for (int j = 0; j < 4; ++j)
                acc[i][j] = wmma_bf16(af[i], bf[j], acc[i][j]);

        if (has_next) {
            *(uint4*)&sA[buf ^ 1][r0][c0] = na0;
            *(uint4*)&sA[buf ^ 1][r1][c1] = na1;
            *(uint4*)&sB[buf ^ 1][r0][c0] = nb0;
            *(uint4*)&sB[buf ^ 1][r1][c1] = nb1;
        }
        __syncthreads();
    }

    const int hi = lane >> 4;
    const int cl = lane & 15;
#pragma unroll
    for (int j = 0; j < 4; ++j) {
        int col = n_blk + wn * 64 + j * 16 + cl;
        if (col < N) {
            float bv = bias[col];
#pragma unroll
            for (int i = 0; i < 2; ++i) {
#pragma unroll
                for (int v = 0; v < 8; ++v) {
                    int row = m_blk + wm * 32 + i * 16 + v + 8 * hi;
                    C[(size_t)row * N + col] = acc[i][j][v] + bv;
                }
            }
        }
    }
}

// ---------- fused LSTM timestep ----------
// g = xg_t + h_src * W_hh^T ; gates i,f,g,o ; update c ; emit h (bf16).
// Wave (blockIdx.x*8 + waveid) owns hidden cols [c0, c0+16) for ALL 16 batch
// rows and all 4 gates, so the nonlinearity is applied in-wave. Register-
// pipelined K loop; h_src = h_all[t-1] (zero buffer at t=0), h written only
// to h_all[t] — stream-ordered launches make that race-free.
__global__ __launch_bounds__(256) void k_lstm_step(const u16* __restrict__ h_src,
                                                   const u16* __restrict__ w_hh,
                                                   const float* __restrict__ xg_t,
                                                   float* __restrict__ c_state,
                                                   u16* __restrict__ h_all_t) {
    const int lane = threadIdx.x & 31;
    const int wave = threadIdx.x >> 5;
    const int c0 = (blockIdx.x * 8 + wave) * 16;

    const u16* arow = h_src + (size_t)(lane & 15) * HIDDEN;
    const u16* brow[4];
#pragma unroll
    for (int g = 0; g < 4; ++g)
        brow[g] = w_hh + (size_t)(g * HIDDEN + c0 + (lane & 15)) * HIDDEN;

    v8f acc[4] = {};
    v16bf a = load_frag(arow, lane);
    v16bf b[4];
#pragma unroll
    for (int g = 0; g < 4; ++g) b[g] = load_frag(brow[g], lane);

    for (int k0 = 0; k0 < HIDDEN; k0 += 32) {
        const int kn = (k0 + 32 < HIDDEN) ? (k0 + 32) : 0;  // wrap: dead loads
        v16bf an = load_frag(arow + kn, lane);
        v16bf bn[4];
#pragma unroll
        for (int g = 0; g < 4; ++g) bn[g] = load_frag(brow[g] + kn, lane);
#pragma unroll
        for (int g = 0; g < 4; ++g) acc[g] = wmma_bf16(a, b[g], acc[g]);
        a = an;
#pragma unroll
        for (int g = 0; g < 4; ++g) b[g] = bn[g];
    }

    const int hi = lane >> 4;
    const int j = c0 + (lane & 15);
#pragma unroll
    for (int v = 0; v < 8; ++v) {
        int bb = v + 8 * hi;
        size_t gbase = (size_t)bb * GATES + j;
        float gi = acc[0][v] + xg_t[gbase];
        float gf = acc[1][v] + xg_t[gbase + HIDDEN];
        float gg = acc[2][v] + xg_t[gbase + 2 * HIDDEN];
        float go = acc[3][v] + xg_t[gbase + 3 * HIDDEN];
        size_t ci = (size_t)bb * HIDDEN + j;
        float cn = sigmoidf(gf) * c_state[ci] + sigmoidf(gi) * tanhf(gg);
        float h  = sigmoidf(go) * tanhf(cn);
        c_state[ci] = cn;
        h_all_t[ci] = f2bf(h);
    }
}

// ---------- in-place row-wise log-softmax over V ----------
__global__ __launch_bounds__(256) void k_log_softmax(float* __restrict__ x, int V) {
    __shared__ float red[256];
    __shared__ float bcast;
    float* row = x + (size_t)blockIdx.x * V;
    int tid = threadIdx.x;

    float m = -3.4e38f;
    for (int i = tid; i < V; i += 256) m = fmaxf(m, row[i]);
    red[tid] = m; __syncthreads();
    for (int s = 128; s > 0; s >>= 1) {
        if (tid < s) red[tid] = fmaxf(red[tid], red[tid + s]);
        __syncthreads();
    }
    if (tid == 0) bcast = red[0];
    __syncthreads();
    m = bcast;
    __syncthreads();

    float sum = 0.0f;
    for (int i = tid; i < V; i += 256) sum += __expf(row[i] - m);
    red[tid] = sum; __syncthreads();
    for (int s = 128; s > 0; s >>= 1) {
        if (tid < s) red[tid] += red[tid + s];
        __syncthreads();
    }
    float ls = __logf(red[0]);
    for (int i = tid; i < V; i += 256) row[i] = row[i] - m - ls;
}

// ---------- host launcher ----------

extern "C" void kernel_launch(void* const* d_in, const int* in_sizes, int n_in,
                              void* d_out, int out_size, void* d_ws, size_t ws_size,
                              hipStream_t stream) {
    (void)in_sizes; (void)n_in; (void)out_size; (void)ws_size;

    const int*   tokens = (const int*)  d_in[0];
    const float* emb    = (const float*)d_in[1];
    const float* w_ih0  = (const float*)d_in[2];
    const float* w_hh0  = (const float*)d_in[3];
    const float* b_ih0  = (const float*)d_in[4];
    const float* b_hh0  = (const float*)d_in[5];
    const float* w_ih1  = (const float*)d_in[6];
    const float* w_hh1  = (const float*)d_in[7];
    const float* b_ih1  = (const float*)d_in[8];
    const float* b_hh1  = (const float*)d_in[9];
    const float* W_out  = (const float*)d_in[10];
    const float* b_out  = (const float*)d_in[11];
    float* out = (float*)d_out;

    // workspace carve-out (256B aligned slots)
    char* base = (char*)d_ws;
    size_t off = 0;
    auto take = [&](size_t bytes) -> char* {
        char* r = base + off;
        off += (bytes + 255) & ~(size_t)255;
        return r;
    };
    u16*   xbf   = (u16*)  take((size_t)MROWS * EMBED * 2);
    u16*   wih0b = (u16*)  take((size_t)GATES * EMBED * 2);
    u16*   whh0b = (u16*)  take((size_t)GATES * HIDDEN * 2);
    u16*   wih1b = (u16*)  take((size_t)GATES * HIDDEN * 2);
    u16*   whh1b = (u16*)  take((size_t)GATES * HIDDEN * 2);
    u16*   woutb = (u16*)  take((size_t)VOCAB * HIDDEN * 2);
    float* xg    = (float*)take((size_t)MROWS * GATES * 4);
    u16*   hall0 = (u16*)  take((size_t)MROWS * HIDDEN * 2);
    u16*   hall1 = (u16*)  take((size_t)MROWS * HIDDEN * 2);
    u16*   hzero = (u16*)  take((size_t)BATCH * HIDDEN * 2);
    float* cst   = (float*)take((size_t)BATCH * HIDDEN * 4);
    float* bias0 = (float*)take((size_t)GATES * 4);
    float* bias1 = (float*)take((size_t)GATES * 4);

    // 1) weight / bias preparation
    const int NW = GATES * HIDDEN;                 // 4M elems
    k_f32_to_bf16<<<4096, 256, 0, stream>>>(w_ih0, wih0b, NW);
    k_f32_to_bf16<<<4096, 256, 0, stream>>>(w_hh0, whh0b, NW);
    k_f32_to_bf16<<<4096, 256, 0, stream>>>(w_ih1, wih1b, NW);
    k_f32_to_bf16<<<4096, 256, 0, stream>>>(w_hh1, whh1b, NW);
    k_f32_to_bf16<<<8192, 256, 0, stream>>>(W_out, woutb, VOCAB * HIDDEN);
    k_bias_combine<<<GATES / 256, 256, 0, stream>>>(b_ih0, b_hh0, b_ih1, b_hh1,
                                                    bias0, bias1);

    // 2) embedding gather (fp32 -> bf16 activations)
    k_embed_gather<<<8192, 256, 0, stream>>>(tokens, emb, xbf, MROWS * EMBED);

    // 3) layer 0: batched input projection, then 128 fused recurrent steps
    k_zero_state<<<(BATCH * HIDDEN) / 256, 256, 0, stream>>>(cst, hzero);
    k_gemm_bf16<<<dim3(GATES / 128, MROWS / 128), 256, 0, stream>>>(
        xbf, wih0b, bias0, xg, GATES, EMBED);
    for (int t = 0; t < SEQ; ++t) {
        const u16* hsrc = (t == 0) ? hzero : (hall0 + (size_t)(t - 1) * BATCH * HIDDEN);
        k_lstm_step<<<8, 256, 0, stream>>>(hsrc, whh0b,
                                           xg + (size_t)t * BATCH * GATES,
                                           cst,
                                           hall0 + (size_t)t * BATCH * HIDDEN);
    }

    // 4) layer 1
    k_zero_state<<<(BATCH * HIDDEN) / 256, 256, 0, stream>>>(cst, hzero);
    k_gemm_bf16<<<dim3(GATES / 128, MROWS / 128), 256, 0, stream>>>(
        hall0, wih1b, bias1, xg, GATES, HIDDEN);
    for (int t = 0; t < SEQ; ++t) {
        const u16* hsrc = (t == 0) ? hzero : (hall1 + (size_t)(t - 1) * BATCH * HIDDEN);
        k_lstm_step<<<8, 256, 0, stream>>>(hsrc, whh1b,
                                           xg + (size_t)t * BATCH * GATES,
                                           cst,
                                           hall1 + (size_t)t * BATCH * HIDDEN);
    }

    // 5) vocab projection straight into d_out, then in-place log-softmax
    k_gemm_bf16<<<dim3((VOCAB + 127) / 128, MROWS / 128), 256, 0, stream>>>(
        hall1, woutb, b_out, out, VOCAB, HIDDEN);
    k_log_softmax<<<MROWS, 256, 0, stream>>>(out, VOCAB);
}
